// CRF_57518202028080
// MI455X (gfx1250) — compile-verified, matrix operand order checked by reading, not run on hardware
//
#include <hip/hip_runtime.h>
#include <hip/hip_bf16.h>
#include <math.h>

#define NB   128
#define BOSL 126
#define EOSL 127

#define L2E  1.4426950408889634f   // log2(e)
#define LN2  0.6931471805599453f   // ln(2)

typedef __attribute__((ext_vector_type(16))) _Float16 v16h;
typedef __attribute__((ext_vector_type(8)))  float    v8f;

union HFrag {
    v16h v;
    _Float16 h[16];
    unsigned int u[8];
};

union HPair {
    _Float16 h[2];
    unsigned int u;
};

__device__ __forceinline__ float fast_exp(float x) {
    return __builtin_amdgcn_exp2f(x * L2E);      // v_exp_f32, no guards
}
__device__ __forceinline__ float fast_log(float x) {
    return LN2 * __builtin_amdgcn_logf(x);       // v_log_f32, no guards
}

// ---------------------------------------------------------------------------
// Partition kernel: one block handles 16 batches; 8 waves, wave w owns the
// 16-label N-tile [16w, 16w+16). expT B-fragments stay resident in VGPRs.
// Per step: rowmax -> exp(f16) -> 4x v_wmma_f32_16x16x32_f16 -> log + emission.
// ---------------------------------------------------------------------------
__global__ __launch_bounds__(256) void crf_partition_kernel(
    const float* __restrict__ emissions,   // [B][T][NB]
    const float* __restrict__ mask,        // [B][T]
    const float* __restrict__ trans,       // [NB][NB]
    float* __restrict__ partition,         // [B]
    int T)
{
    const int SA = 132;                    // alpha row stride (floats), padded
    const int SE = 136;                    // expA row stride (halves), padded
    __shared__ float    alpha[16 * 132];
    __shared__ _Float16 expA [16 * 136];
    __shared__ float    mx[16];
    __shared__ float    mkS[16];

    const int tid  = threadIdx.x;
    const int wave = tid >> 5;
    const int lane = tid & 31;
    const int g    = lane >> 4;            // half-wave group
    const int ln   = lane & 15;
    const int b0   = blockIdx.x * 16;

    // alpha0[b][j] = T[BOS][j] + e[b][0][j]
    for (int i = tid; i < 16 * NB; i += 256) {
        int b = i >> 7, j = i & 127;
        alpha[b * SA + j] = trans[BOSL * NB + j] +
                            emissions[(size_t)(b0 + b) * T * NB + j];
    }

    // Resident B fragments: expT for this wave's N-tile, 4 K-chunks of 32.
    // B fragment layout mirrors the 16-bit A 16x32 layout (lane%16 = N).
    HFrag bfrag[4];
    {
        const int n = wave * 16 + ln;
        #pragma unroll
        for (int c = 0; c < 4; ++c) {
            #pragma unroll
            for (int h = 0; h < 16; ++h) {
                int k = 32 * c + ((h < 8) ? (g * 8 + h) : (8 + g * 8 + h));
                bfrag[c].h[h] = (_Float16)fast_exp(trans[k * NB + n]);
            }
        }
    }
    __syncthreads();

    const int rowb = tid >> 4;             // 0..15 : batch row for reductions
    const int s    = tid & 15;             // 16 lanes per row

    for (int t = 1; t < T; ++t) {
        // ---- row max over labels; stage mask[.,t] --------------------
        float m = -INFINITY;
        #pragma unroll
        for (int k = 0; k < 8; ++k)
            m = fmaxf(m, alpha[rowb * SA + s + 16 * k]);
        #pragma unroll
        for (int d = 1; d < 16; d <<= 1)
            m = fmaxf(m, __shfl_xor(m, d, 16));
        if (s == 0) mx[rowb] = m;
        if (tid < 16) mkS[tid] = mask[(size_t)(b0 + tid) * T + t];
        __syncthreads();

        // ---- expA[b][j] = exp(alpha - rowmax) as packed f16 pairs ----
        for (int i = tid; i < 16 * (NB / 2); i += 256) {
            int b  = i >> 6;
            int j  = (i & 63) * 2;
            float mb = mx[b];
            float a0 = alpha[b * SA + j];
            float a1 = alpha[b * SA + j + 1];
            HPair p;
            p.h[0] = (_Float16)fast_exp(a0 - mb);
            p.h[1] = (_Float16)fast_exp(a1 - mb);
            *(unsigned int*)&expA[b * SE + j] = p.u;
        }
        __syncthreads();

        // ---- WMMA: acc = expA(16x128) * expT(128x16-tile) ------------
        v8f acc = {0.f, 0.f, 0.f, 0.f, 0.f, 0.f, 0.f, 0.f};
        #pragma unroll
        for (int c = 0; c < 4; ++c) {
            HFrag af;
            const int kb = 32 * c + g * 8;
            #pragma unroll
            for (int j = 0; j < 8; ++j) {
                int k = kb + 2 * j + ((j < 4) ? 0 : 8);
                af.u[j] = *(const unsigned int*)&expA[ln * SE + k];
            }
            acc = __builtin_amdgcn_wmma_f32_16x16x32_f16(
                      false, af.v, false, bfrag[c].v,
                      (short)0, acc, false, false);
        }

        // ---- epilogue: alpha' = rowmax + ln(acc) + e_t, mask blend ---
        float nv[8];
        const int j = wave * 16 + ln;
        #pragma unroll
        for (int r = 0; r < 8; ++r) {
            const int b = r + 8 * g;
            float e  = emissions[(size_t)(b0 + b) * T * NB + (size_t)t * NB + j];
            float na = mx[b] + fast_log(fmaxf(acc[r], 1e-30f)) + e;
            float mk = mkS[b];
            nv[r] = mk * na + (1.f - mk) * alpha[b * SA + j];
        }
        #pragma unroll
        for (int r = 0; r < 8; ++r) {
            const int b = r + 8 * g;
            alpha[b * SA + j] = nv[r];
        }
        __syncthreads();
    }

    // ---- partition[b] = logsumexp_j(alpha[b][j] + T[j][EOS]) ----------
    {
        float vals[8];
        float m = -INFINITY;
        #pragma unroll
        for (int k = 0; k < 8; ++k) {
            int j = s + 16 * k;
            vals[k] = alpha[rowb * SA + j] + trans[j * NB + EOSL];
            m = fmaxf(m, vals[k]);
        }
        #pragma unroll
        for (int d = 1; d < 16; d <<= 1)
            m = fmaxf(m, __shfl_xor(m, d, 16));
        float sum = 0.f;
        #pragma unroll
        for (int k = 0; k < 8; ++k)
            sum += fast_exp(vals[k] - m);
        #pragma unroll
        for (int d = 1; d < 16; d <<= 1)
            sum += __shfl_xor(sum, d, 16);
        if (s == 0) partition[b0 + rowb] = m + fast_log(sum);
    }
}

// ---------------------------------------------------------------------------
// Scores kernel: one block per batch, gather + masked sum over T.
// ---------------------------------------------------------------------------
__global__ __launch_bounds__(256) void crf_score_kernel(
    const float* __restrict__ emissions,
    const int*   __restrict__ tags,       // [B][T]
    const float* __restrict__ mask,       // [B][T]
    const float* __restrict__ trans,
    float* __restrict__ scores,           // [B]
    int T)
{
    __shared__ float red[256];
    __shared__ float redm[256];
    const int b   = blockIdx.x;
    const int tid = threadIdx.x;
    const size_t be = (size_t)b * T * NB;
    const size_t bt = (size_t)b * T;

    float acc = 0.f, msum = 0.f;
    for (int t = tid; t < T; t += 256) {
        float mk = mask[bt + t];
        msum += mk;
        if (t == 0) {
            int tg0 = tags[bt];
            acc += trans[BOSL * NB + tg0] + emissions[be + tg0];
        } else {
            int tgp = tags[bt + t - 1];
            int tg  = tags[bt + t];
            acc += mk * (emissions[be + (size_t)t * NB + tg] + trans[tgp * NB + tg]);
        }
    }
    red[tid]  = acc;
    redm[tid] = msum;
    __syncthreads();
    for (int sft = 128; sft > 0; sft >>= 1) {
        if (tid < sft) {
            red[tid]  += red[tid + sft];
            redm[tid] += redm[tid + sft];
        }
        __syncthreads();
    }
    if (tid == 0) {
        int last = (int)redm[0] - 1;
        if (last < 0) last = 0;
        if (last > T - 1) last = T - 1;
        int lt = tags[bt + last];
        scores[b] = red[0] + trans[lt * NB + EOSL];
    }
}

// ---------------------------------------------------------------------------
// Final: loss = -mean(scores - partition)
// ---------------------------------------------------------------------------
__global__ void crf_final_kernel(const float* __restrict__ scores,
                                 const float* __restrict__ partition,
                                 float* __restrict__ out, int B)
{
    __shared__ float red[128];
    const int tid = threadIdx.x;
    red[tid] = scores[tid] - partition[tid];
    __syncthreads();
    for (int sft = 64; sft > 0; sft >>= 1) {
        if (tid < sft) red[tid] += red[tid + sft];
        __syncthreads();
    }
    if (tid == 0) out[0] = -red[0] / (float)B;
}

// ---------------------------------------------------------------------------
extern "C" void kernel_launch(void* const* d_in, const int* in_sizes, int n_in,
                              void* d_out, int out_size, void* d_ws, size_t ws_size,
                              hipStream_t stream)
{
    const float* emissions = (const float*)d_in[0];   // [B][T][128] f32
    const int*   tags      = (const int*)  d_in[1];   // [B][T] int32
    const float* mask      = (const float*)d_in[2];   // [B][T] f32
    const float* trans     = (const float*)d_in[3];   // [128][128] f32

    const int B = 128;
    const int T = in_sizes[1] / B;                    // B*T elements in tags

    float* scores    = (float*)d_ws;                  // [B]
    float* partition = scores + B;                    // [B]

    crf_score_kernel<<<B, 256, 0, stream>>>(emissions, tags, mask, trans, scores, T);
    crf_partition_kernel<<<B / 16, 256, 0, stream>>>(emissions, mask, trans, partition, T);
    crf_final_kernel<<<1, B, 0, stream>>>(scores, partition, (float*)d_out, B);
}
